// GraphNet_72541997629574
// MI455X (gfx1250) — matrix operand order repaired and checked by reading, not compiled, not run on hardware
//
#include <hip/hip_runtime.h>
#include <hip/hip_bf16.h>
#include <math.h>

// ---------------------------------------------------------------------------
// GATv2 GNN pipeline for gfx1250 (MI455X).
// Dense 128x128 linears run on v_wmma_f32_16x16x32_bf16 (bf16 in, f32 acc).
// Edge-stream fusions:
//   - edge MLP layer-0 (rank-1 in scalar edge_attr) fused into the eW1 GEMM's
//     A-fragment generation (saves 256MB HBM round trip).
//   - attention-logit pass fused into the e@We GEMM epilogue via half-wave
//     shuffle reduction (col tile == head), eliminating the ee buffer.
// Segment softmax + weighted scatter use global atomics.
// ---------------------------------------------------------------------------

typedef __attribute__((ext_vector_type(16))) __bf16 v16bf;
typedef __attribute__((ext_vector_type(8)))  __bf16 v8bf;
typedef __attribute__((ext_vector_type(8)))  float  v8f;

#define HID      128
#define NNODES   16384
#define NEDGES   524288
#define NHEAD    8
#define CHAN     16
#define NEG_SLOPE 0.2f

// order-preserving float<->uint encoding for atomicMax on signed floats
__device__ __forceinline__ unsigned enc_f(float x) {
  unsigned u = __float_as_uint(x);
  return u ^ ((u >> 31) ? 0xFFFFFFFFu : 0x80000000u);
}
__device__ __forceinline__ float dec_f(unsigned u) {
  return __uint_as_float(u ^ ((u & 0x80000000u) ? 0x80000000u : 0xFFFFFFFFu));
}

// --- fragment helpers (ISA 16-bit WMMA operand layouts, wave32) -------------
// A (16x32, row-major source): lane lm row, half-wave K split {0-7,16-23}/{8-15,24-31}
__device__ __forceinline__ v16bf load_afrag(const __bf16* arow, int k, int hiA) {
  v8bf lo = *(const v8bf*)(arow + k * 32 + hiA);
  v8bf hh = *(const v8bf*)(arow + k * 32 + 16 + hiA);
  v16bf f;
#pragma unroll
  for (int i = 0; i < 8; ++i) { f[i] = lo[i]; f[i + 8] = hh[i]; }
  return f;
}
// B (32x16, from transposed [out][in] weights): lane lm col, half-wave K split {0-7,16-23}/{8-15,24-31}
__device__ __forceinline__ v16bf load_bfrag(const __bf16* wrow, int k, int hiB) {
  v8bf lo = *(const v8bf*)(wrow + k * 32 + (hiB >> 1));
  v8bf hh = *(const v8bf*)(wrow + k * 32 + 16 + (hiB >> 1));
  v16bf f;
#pragma unroll
  for (int i = 0; i < 8; ++i) { f[i] = lo[i]; f[i + 8] = hh[i]; }
  return f;
}

// --- weight prep: fp32 [in][out] -> bf16 transposed [out][in] ---------------
__global__ void prep_wt(const float* __restrict__ W, __bf16* __restrict__ Wt) {
  int t = blockIdx.x * blockDim.x + threadIdx.x;        // 128*128 threads
  int o = t >> 7, i = t & 127;
  Wt[o * HID + i] = (__bf16)W[i * HID + o];
}

// --- node MLP first linear: [N,2] @ [2,128] + b, relu, -> bf16 --------------
__global__ void node0(const float* __restrict__ x, const float* __restrict__ W0,
                      const float* __restrict__ b0, __bf16* __restrict__ out) {
  int t = blockIdx.x * blockDim.x + threadIdx.x;
  if (t >= NNODES * HID) return;
  int n = t >> 7, c = t & 127;
  float v = x[2 * n] * W0[c] + x[2 * n + 1] * W0[HID + c] + b0[c];
  out[t] = (__bf16)fmaxf(v, 0.f);
}

// --- WMMA GEMM: C[M,128] = A[M,128](bf16) @ Wt[128,128](bf16,[out][in]) + b -
// 256 threads = 8 waves; wave w owns rows [blk*128 + w*16, +16).
// Output type is a compile-time choice -> unconditional stores, no branches.
template <bool BF16OUT>
__global__ void gemm128(const __bf16* __restrict__ A, const __bf16* __restrict__ Wt,
                        const float* __restrict__ bias, void* __restrict__ Cout, int M) {
  int wave = threadIdx.x >> 5;
  int lane = threadIdx.x & 31;
  int row0 = blockIdx.x * 128 + wave * 16;
  if (row0 >= M) return;                       // wave-uniform; M is mult of 128 here
  int lm  = lane & 15;
  int hiA = (lane >= 16) ? 8 : 0;
  int hiB = (lane >= 16) ? 16 : 0;
  int rofs = (lane >= 16) ? 8 : 0;

  const __bf16* arow = A + (size_t)(row0 + lm) * HID;
  v16bf afrag[4];
#pragma unroll
  for (int k = 0; k < 4; ++k) afrag[k] = load_afrag(arow, k, hiA);

#pragma unroll
  for (int nt = 0; nt < 8; ++nt) {
    const __bf16* wrow = Wt + (size_t)(nt * 16 + lm) * HID;
    v8f acc = {};
#pragma unroll
    for (int k = 0; k < 4; ++k) {
      v16bf bfrag = load_bfrag(wrow, k, hiB);
      acc = __builtin_amdgcn_wmma_f32_16x16x32_bf16(
          false, afrag[k], false, bfrag, (short)0, acc, false, false);
    }
    int col = nt * 16 + lm;
    float bv = bias[col];
#pragma unroll
    for (int v = 0; v < 8; ++v) {
      size_t idx = (size_t)(row0 + v + rofs) * HID + col;
      float val = acc[v] + bv;
      if constexpr (BF16OUT) ((__bf16*)Cout)[idx] = (__bf16)val;
      else                   ((float*)Cout)[idx]  = val;
    }
  }
}

// --- fused edge MLP: e = relu(ea*eW0+eb0) @ eW1 + eb1, A built in registers -
__global__ void gemm_edge_e(const float* __restrict__ ea, const float* __restrict__ eW0,
                            const float* __restrict__ eb0, const __bf16* __restrict__ Wt,
                            const float* __restrict__ bias, __bf16* __restrict__ Cb) {
  int wave = threadIdx.x >> 5;
  int lane = threadIdx.x & 31;
  int row0 = blockIdx.x * 128 + wave * 16;
  int lm  = lane & 15;
  int hiA = (lane >= 16) ? 8 : 0;
  int hiB = (lane >= 16) ? 16 : 0;
  int rofs = (lane >= 16) ? 8 : 0;

  float a = ea[row0 + lm];
  v16bf afrag[4];
#pragma unroll
  for (int k = 0; k < 4; ++k) {
#pragma unroll
    for (int i = 0; i < 8; ++i) {
      int K1 = k * 32 + hiA + i;
      int K2 = k * 32 + 16 + hiA + i;
      afrag[k][i]     = (__bf16)fmaxf(a * eW0[K1] + eb0[K1], 0.f);
      afrag[k][i + 8] = (__bf16)fmaxf(a * eW0[K2] + eb0[K2], 0.f);
    }
  }

#pragma unroll
  for (int nt = 0; nt < 8; ++nt) {
    const __bf16* wrow = Wt + (size_t)(nt * 16 + lm) * HID;
    v8f acc = {};
#pragma unroll
    for (int k = 0; k < 4; ++k) {
      v16bf bfrag = load_bfrag(wrow, k, hiB);
      acc = __builtin_amdgcn_wmma_f32_16x16x32_bf16(
          false, afrag[k], false, bfrag, (short)0, acc, false, false);
    }
    int col = nt * 16 + lm;
    float bv = bias[col];
#pragma unroll
    for (int v = 0; v < 8; ++v)
      Cb[(size_t)(row0 + v + rofs) * HID + col] = (__bf16)(acc[v] + bv);
  }
}

// --- fused ee GEMM + attention pass 1 ---------------------------------------
// ee = e @ We (no bias); col tile nt == head nt. Epilogue forms
// m = leakyrelu(ee + xl[src] + xr[dst]), logit = sum_c m*att via half-wave
// shuffle reduction; lane lm==0 writes logit and atomic-maxes per (dst,head).
__global__ void gemm_attn_logit(const __bf16* __restrict__ A, const __bf16* __restrict__ Wt,
                                const int* __restrict__ src, const int* __restrict__ dst,
                                const float* __restrict__ xl, const float* __restrict__ xr,
                                const float* __restrict__ att,
                                float* __restrict__ logits, unsigned* __restrict__ amax) {
  int wave = threadIdx.x >> 5;
  int lane = threadIdx.x & 31;
  int row0 = blockIdx.x * 128 + wave * 16;   // 128 edges per block
  int lm  = lane & 15;
  int hiA = (lane >= 16) ? 8 : 0;
  int hiB = (lane >= 16) ? 16 : 0;
  int rofs = (lane >= 16) ? 8 : 0;

  const __bf16* arow = A + (size_t)(row0 + lm) * HID;
  v16bf afrag[4];
#pragma unroll
  for (int k = 0; k < 4; ++k) afrag[k] = load_afrag(arow, k, hiA);

#pragma unroll
  for (int nt = 0; nt < 8; ++nt) {           // nt == head index
    const __bf16* wrow = Wt + (size_t)(nt * 16 + lm) * HID;
    v8f acc = {};
#pragma unroll
    for (int k = 0; k < 4; ++k) {
      v16bf bfrag = load_bfrag(wrow, k, hiB);
      acc = __builtin_amdgcn_wmma_f32_16x16x32_bf16(
          false, afrag[k], false, bfrag, (short)0, acc, false, false);
    }
    int col = nt * 16 + lm;                  // channel within head nt
    float attv = att[col];
#pragma unroll
    for (int v = 0; v < 8; ++v) {
      int row = row0 + v + rofs;             // edge id
      int s = src[row], d = dst[row];
      float m = acc[v] + xl[(size_t)s * HID + col] + xr[(size_t)d * HID + col];
      m = (m > 0.f) ? m : (NEG_SLOPE * m);
      float c = m * attv;
      c += __shfl_xor(c, 1, 32);             // reduce over 16 channels (half-wave)
      c += __shfl_xor(c, 2, 32);
      c += __shfl_xor(c, 4, 32);
      c += __shfl_xor(c, 8, 32);
      if (lm == 0) {
        logits[(size_t)row * NHEAD + nt] = c;
        atomicMax(amax + (size_t)d * NHEAD + nt, enc_f(c));
      }
    }
  }
}

// --- per-layer scratch reset ------------------------------------------------
__global__ void init_layer(float* __restrict__ acc, unsigned* __restrict__ amax,
                           float* __restrict__ ssum) {
  int t = blockIdx.x * blockDim.x + threadIdx.x;
  if (t < NNODES * HID) acc[t] = 0.f;
  if (t < NNODES * NHEAD) { amax[t] = 0u; ssum[t] = 0.f; }
}

// --- attention pass 2: exp + segment sum ------------------------------------
__global__ void attn_exp(const int* __restrict__ dst, float* __restrict__ buf,
                         const unsigned* __restrict__ amax, float* __restrict__ ssum) {
  int t = blockIdx.x * blockDim.x + threadIdx.x;
  if (t >= NEDGES * NHEAD) return;
  int e = t >> 3, h = t & 7;
  int d = dst[e];
  float ex = __expf(buf[t] - dec_f(amax[(size_t)d * NHEAD + h]));
  buf[t] = ex;
  atomicAdd(ssum + (size_t)d * NHEAD + h, ex);
}

// --- attention pass 3: alpha-weighted scatter of xl[src] --------------------
__global__ void attn_agg(const int* __restrict__ src, const int* __restrict__ dst,
                         const float* __restrict__ buf, const float* __restrict__ ssum,
                         const float* __restrict__ xl, float* __restrict__ acc) {
  int t = blockIdx.x * blockDim.x + threadIdx.x;
  if (t >= NEDGES * NHEAD) return;
  int e = t >> 3, h = t & 7;
  int s = src[e], d = dst[e];
  float alpha = buf[t] / (ssum[(size_t)d * NHEAD + h] + 1e-16f);
  const float* pl = xl + (size_t)s * HID + h * CHAN;
  float*       pa = acc + (size_t)d * HID + h * CHAN;
#pragma unroll
  for (int c = 0; c < CHAN; ++c) atomicAdd(pa + c, pl[c] * alpha);
}

// --- epilogue: h = relu(acc + bias) -> bf16 ---------------------------------
__global__ void post_relu(const float* __restrict__ acc, const float* __restrict__ bias,
                          __bf16* __restrict__ hout) {
  int t = blockIdx.x * blockDim.x + threadIdx.x;
  if (t >= NNODES * HID) return;
  int c = t & 127;
  hout[t] = (__bf16)fmaxf(acc[t] + bias[c], 0.f);
}

// ---------------------------------------------------------------------------
extern "C" void kernel_launch(void* const* d_in, const int* in_sizes, int n_in,
                              void* d_out, int out_size, void* d_ws, size_t ws_size,
                              hipStream_t stream) {
  (void)in_sizes; (void)n_in; (void)out_size; (void)ws_size;
  // Input order = setup_inputs() dict order, params flattened in insertion order.
  const float* x   = (const float*)d_in[0];
  const float* ea  = (const float*)d_in[1];
  const float* nW0 = (const float*)d_in[2];
  const float* nb0 = (const float*)d_in[3];
  const float* nW1 = (const float*)d_in[4];
  const float* nb1 = (const float*)d_in[5];
  const float* eW0 = (const float*)d_in[6];
  const float* eb0 = (const float*)d_in[7];
  const float* eW1 = (const float*)d_in[8];
  const float* eb1 = (const float*)d_in[9];
  // layers l: base 10 + 7*l : Wl, bl, Wr, br, We, att, bias
  const float* fW  = (const float*)d_in[31];
  const float* fb  = (const float*)d_in[32];
  const int*   eidx = (const int*)d_in[33];
  const int*   src = eidx;
  const int*   dst = eidx + NEDGES;

  // ---- workspace layout (256B aligned) ----
  char* p = (char*)d_ws;
  auto take = [&](size_t bytes) { char* r = p; p += (bytes + 255) & ~(size_t)255; return r; };
  __bf16* wt    = (__bf16*)take((size_t)12 * HID * HID * sizeof(__bf16)); // prepped weights
  __bf16* h_bf  = (__bf16*)take((size_t)NNODES * HID * sizeof(__bf16));   // node features
  __bf16* e_bf  = (__bf16*)take((size_t)NEDGES * HID * sizeof(__bf16));   // edge features e
  __bf16* stage = (__bf16*)take((size_t)NNODES * HID * sizeof(__bf16));   // node MLP staging
  float*  xl_f  = (float*) take((size_t)NNODES * HID * sizeof(float));
  float*  xr_f  = (float*) take((size_t)NNODES * HID * sizeof(float));
  float*  acc_f = (float*) take((size_t)NNODES * HID * sizeof(float));
  float*  ebuf  = (float*) take((size_t)NEDGES * NHEAD * sizeof(float));  // logits -> exp
  unsigned* amax = (unsigned*)take((size_t)NNODES * NHEAD * sizeof(unsigned));
  float*  ssum  = (float*) take((size_t)NNODES * NHEAD * sizeof(float));

  const int T = 256;
  dim3 blk(T);
  const int gW  = (HID * HID) / T;          // 64
  const int gN  = (NNODES * HID) / T;       // 8192
  const int gEH = (NEDGES * NHEAD) / T;     // 16384
  const int gmN = NNODES / 128;             // 128 GEMM blocks
  const int gmE = NEDGES / 128;             // 4096 GEMM blocks

  // ---- prep all 128x128 weights: transpose + bf16 ----
  // slots: 0=nW1, 1=eW1, 2+3l={Wl,Wr,We}, 11=fW
  prep_wt<<<gW, blk, 0, stream>>>(nW1, wt + 0 * HID * HID);
  prep_wt<<<gW, blk, 0, stream>>>(eW1, wt + 1 * HID * HID);
  for (int l = 0; l < 3; ++l) {
    const float* Wl = (const float*)d_in[10 + 7 * l + 0];
    const float* Wr = (const float*)d_in[10 + 7 * l + 2];
    const float* We = (const float*)d_in[10 + 7 * l + 4];
    prep_wt<<<gW, blk, 0, stream>>>(Wl, wt + (size_t)(2 + 3 * l) * HID * HID);
    prep_wt<<<gW, blk, 0, stream>>>(Wr, wt + (size_t)(3 + 3 * l) * HID * HID);
    prep_wt<<<gW, blk, 0, stream>>>(We, wt + (size_t)(4 + 3 * l) * HID * HID);
  }
  prep_wt<<<gW, blk, 0, stream>>>(fW, wt + (size_t)11 * HID * HID);

  // ---- node MLP: h = relu(x@nW0+nb0) @ nW1 + nb1 ----
  node0<<<gN, blk, 0, stream>>>(x, nW0, nb0, stage);
  gemm128<true><<<gmN, blk, 0, stream>>>(stage, wt + 0 * HID * HID, nb1, h_bf, NNODES);

  // ---- edge MLP fused: e = relu(ea*eW0+eb0) @ eW1 + eb1 ----
  gemm_edge_e<<<gmE, blk, 0, stream>>>(ea, eW0, eb0, wt + 1 * HID * HID, eb1, e_bf);

  // ---- 3 GATv2 layers ----
  for (int l = 0; l < 3; ++l) {
    const float* bl   = (const float*)d_in[10 + 7 * l + 1];
    const float* br   = (const float*)d_in[10 + 7 * l + 3];
    const float* att  = (const float*)d_in[10 + 7 * l + 5];
    const float* bias = (const float*)d_in[10 + 7 * l + 6];

    gemm128<false><<<gmN, blk, 0, stream>>>(h_bf, wt + (size_t)(2 + 3 * l) * HID * HID,
                                            bl, xl_f, NNODES);
    gemm128<false><<<gmN, blk, 0, stream>>>(h_bf, wt + (size_t)(3 + 3 * l) * HID * HID,
                                            br, xr_f, NNODES);
    init_layer<<<gN, blk, 0, stream>>>(acc_f, amax, ssum);
    gemm_attn_logit<<<gmE, blk, 0, stream>>>(e_bf, wt + (size_t)(4 + 3 * l) * HID * HID,
                                             src, dst, xl_f, xr_f, att, ebuf, amax);
    attn_exp<<<gEH, blk, 0, stream>>>(dst, ebuf, amax, ssum);
    attn_agg<<<gEH, blk, 0, stream>>>(src, dst, ebuf, ssum, xl_f, acc_f);
    post_relu<<<gN, blk, 0, stream>>>(acc_f, bias, h_bf);
  }

  // ---- final linear -> d_out (to_dense_batch is an identity reshape) ----
  gemm128<false><<<gmN, blk, 0, stream>>>(h_bf, wt + (size_t)11 * HID * HID,
                                          fb, d_out, NNODES);
}
